// DGCNN_cls_semseg_27264452395467
// MI455X (gfx1250) — compile-verified, matrix operand order checked by reading, not run on hardware
//
#include <hip/hip_runtime.h>
#include <hip/hip_bf16.h>
#include <math.h>

typedef __attribute__((ext_vector_type(16))) __bf16 v16bf;
typedef __attribute__((ext_vector_type(8)))  float  v8f;

#define NB 8
#define NPTS 2048
#define KNN 20
#define NEGS 0.2f

__device__ __forceinline__ float4 ld4(const float* p) { return *(const float4*)p; }
__device__ __forceinline__ float4 sub4(float4 a, float4 b) {
    return make_float4(a.x - b.x, a.y - b.y, a.z - b.z, a.w - b.w);
}
// Pack two contiguous 8-float runs into the 16-bit A/B fragment layout
// (e=0..7 -> K=run0+e, e=8..15 -> K=run1+(e-8); CDNA5 ISA 7.12.2).
__device__ __forceinline__ v16bf pack16(float4 a0, float4 a1, float4 a2, float4 a3) {
    v16bf v;
    v[0]  = (__bf16)a0.x; v[1]  = (__bf16)a0.y; v[2]  = (__bf16)a0.z; v[3]  = (__bf16)a0.w;
    v[4]  = (__bf16)a1.x; v[5]  = (__bf16)a1.y; v[6]  = (__bf16)a1.z; v[7]  = (__bf16)a1.w;
    v[8]  = (__bf16)a2.x; v[9]  = (__bf16)a2.y; v[10] = (__bf16)a2.z; v[11] = (__bf16)a2.w;
    v[12] = (__bf16)a3.x; v[13] = (__bf16)a3.y; v[14] = (__bf16)a3.z; v[15] = (__bf16)a3.w;
    return v;
}

// =====================  transpose x [B,3,N] -> xt [B,N,3]  =====================
__global__ void transpose_kernel(const float* __restrict__ x, float* __restrict__ xt) {
    int i = blockIdx.x * blockDim.x + threadIdx.x;
    if (i >= NB * NPTS * 3) return;
    int c = i % 3; int n = (i / 3) % NPTS; int b = i / (3 * NPTS);
    xt[i] = x[(size_t)b * 3 * NPTS + (size_t)c * NPTS + n];
}

// =====================  kNN (top-20 by -||xi-xj||^2), LDS-tiled  ================
template<int C>
__global__ void knn_kernel(const float* __restrict__ pts, int* __restrict__ idx_out) {
    const int TILE = 128;
    __shared__ float sh[TILE * C];
    __shared__ float shsq[TILE];
    int b = blockIdx.y;
    int q = blockIdx.x * blockDim.x + threadIdx.x;

    float qv[C];
#pragma unroll
    for (int c = 0; c < C; ++c) qv[c] = pts[((size_t)b * NPTS + q) * C + c];

    float best[KNN]; int bidx[KNN];
#pragma unroll
    for (int t = 0; t < KNN; ++t) { best[t] = -1e30f; bidx[t] = 0; }

    for (int j0 = 0; j0 < NPTS; j0 += TILE) {
        __syncthreads();
        for (int t = threadIdx.x; t < TILE; t += blockDim.x) {
            float sq = 0.f;
            for (int c = 0; c < C; ++c) {
                float v = pts[((size_t)b * NPTS + j0 + t) * C + c];
                sh[t * C + c] = v; sq += v * v;
            }
            shsq[t] = sq;
        }
        __syncthreads();
        for (int t = 0; t < TILE; ++t) {
            float dot = 0.f;
#pragma unroll
            for (int c = 0; c < C; ++c) dot += qv[c] * sh[t * C + c];
            float sc = 2.f * dot - shsq[t];  // rank-equivalent to pd
            if (sc > best[KNN - 1]) {
                best[KNN - 1] = sc; bidx[KNN - 1] = j0 + t;
#pragma unroll
                for (int u = KNN - 1; u > 0; --u) {
                    if (best[u] > best[u - 1]) {
                        float tv = best[u]; best[u] = best[u - 1]; best[u - 1] = tv;
                        int ti = bidx[u]; bidx[u] = bidx[u - 1]; bidx[u - 1] = ti;
                    }
                }
            }
        }
    }
#pragma unroll
    for (int t = 0; t < KNN; ++t)
        idx_out[((size_t)b * NPTS + q) * KNN + t] = bidx[t];
}

// ==========  WMMA GEMM: out[M,COUT] = leaky((A[M,CIN] x W^T)*s+b)  ==========
// One wave computes a 16x(16*NT) tile; A fragment shared across NT sub-tiles.
template<int CIN, int COUT, int NT>
__global__ void gemm_kernel(const float* __restrict__ A, const float* __restrict__ W,
                            const float* __restrict__ sc, const float* __restrict__ bi,
                            float* __restrict__ out, int M) {
    int wave = blockIdx.x * (blockDim.x >> 5) + (threadIdx.x >> 5);
    int lane = threadIdx.x & 31;
    constexpr int tilesN = COUT / (16 * NT);
    int tiles = (M >> 4) * tilesN;
    if (wave >= tiles) return;
    int tm = wave / tilesN, tg = wave % tilesN;
    int row0 = tm << 4, col0 = tg * 16 * NT;
    int half = lane >> 4, mr = lane & 15;

    const float* arow = A + (size_t)(row0 + mr) * CIN;
    const float* wrow[NT];
#pragma unroll
    for (int t = 0; t < NT; ++t) wrow[t] = W + (size_t)(col0 + t * 16 + mr) * CIN;

    v8f acc[NT];
#pragma unroll
    for (int t = 0; t < NT; ++t) acc[t] = (v8f){};

    for (int k0 = 0; k0 < CIN; k0 += 32) {
        int r0 = k0 + half * 8;
        int r1 = k0 + 16 + half * 8;
        v16bf af = pack16(ld4(arow + r0), ld4(arow + r0 + 4),
                          ld4(arow + r1), ld4(arow + r1 + 4));
#pragma unroll
        for (int t = 0; t < NT; ++t) {
            v16bf bfv = pack16(ld4(wrow[t] + r0), ld4(wrow[t] + r0 + 4),
                               ld4(wrow[t] + r1), ld4(wrow[t] + r1 + 4));
            acc[t] = __builtin_amdgcn_wmma_f32_16x16x32_bf16(false, af, false, bfv,
                                                             (short)0, acc[t], false, false);
        }
    }
    int mb = row0 + half * 8;
#pragma unroll
    for (int t = 0; t < NT; ++t) {
        int nc = col0 + t * 16 + (lane & 15);
        float s_ = sc[nc], b_ = bi[nc];
#pragma unroll
        for (int r = 0; r < 8; ++r) {
            float z = acc[t][r] * s_ + b_;
            z = z > 0.f ? z : NEGS * z;
            out[(size_t)(mb + r) * COUT + nc] = z;
        }
    }
}

// ==========  WMMA edge GEMM (C multiple of 32): a[r,c] = c<C ? nbr-ctr : ctr  ==========
template<int C, int COUT, int NT>
__global__ void edge_gemm_kernel(const float* __restrict__ src, const int* __restrict__ idxp,
                                 const float* __restrict__ W, const float* __restrict__ sc,
                                 const float* __restrict__ bi, float* __restrict__ out,
                                 int M) {
    constexpr int CIN = 2 * C;
    int wave = blockIdx.x * (blockDim.x >> 5) + (threadIdx.x >> 5);
    int lane = threadIdx.x & 31;
    constexpr int tilesN = COUT / (16 * NT);
    int tiles = (M >> 4) * tilesN;
    if (wave >= tiles) return;
    int tm = wave / tilesN, tg = wave % tilesN;
    int row0 = tm << 4, col0 = tg * 16 * NT;
    int half = lane >> 4, mr = lane & 15;

    int r = row0 + mr;                 // r = (b*N + n)*KNN + j
    int bn = r / KNN;
    int n = bn % NPTS, b = bn / NPTS;
    int nb = idxp[r];
    const float* ctr = src + ((size_t)b * NPTS + n) * C;
    const float* nbr = src + ((size_t)b * NPTS + nb) * C;
    const float* wrow[NT];
#pragma unroll
    for (int t = 0; t < NT; ++t) wrow[t] = W + (size_t)(col0 + t * 16 + mr) * CIN;

    v8f acc[NT];
#pragma unroll
    for (int t = 0; t < NT; ++t) acc[t] = (v8f){};

#pragma unroll
    for (int k0 = 0; k0 < CIN; k0 += 32) {
        int r0 = k0 + half * 8;
        int r1 = k0 + 16 + half * 8;
        float4 e0, e1, e2, e3;
        if (k0 < C) {            // runs never straddle the C boundary (C % 32 == 0)
            e0 = sub4(ld4(nbr + r0), ld4(ctr + r0));
            e1 = sub4(ld4(nbr + r0 + 4), ld4(ctr + r0 + 4));
        } else {
            e0 = ld4(ctr + r0 - C);
            e1 = ld4(ctr + r0 - C + 4);
        }
        if (k0 + 16 < C) {
            e2 = sub4(ld4(nbr + r1), ld4(ctr + r1));
            e3 = sub4(ld4(nbr + r1 + 4), ld4(ctr + r1 + 4));
        } else {
            e2 = ld4(ctr + r1 - C);
            e3 = ld4(ctr + r1 - C + 4);
        }
        v16bf af = pack16(e0, e1, e2, e3);
#pragma unroll
        for (int t = 0; t < NT; ++t) {
            v16bf bfv = pack16(ld4(wrow[t] + r0), ld4(wrow[t] + r0 + 4),
                               ld4(wrow[t] + r1), ld4(wrow[t] + r1 + 4));
            acc[t] = __builtin_amdgcn_wmma_f32_16x16x32_bf16(false, af, false, bfv,
                                                             (short)0, acc[t], false, false);
        }
    }
    int mb = row0 + half * 8;
#pragma unroll
    for (int t = 0; t < NT; ++t) {
        int nc = col0 + t * 16 + (lane & 15);
        float s_ = sc[nc], b_ = bi[nc];
#pragma unroll
        for (int rr = 0; rr < 8; ++rr) {
            float z = acc[t][rr] * s_ + b_;
            z = z > 0.f ? z : NEGS * z;
            out[(size_t)(mb + rr) * COUT + nc] = z;
        }
    }
}

// ==========  WMMA edge GEMM for C=3 (CIN=6): register-select fragment  ==========
template<int COUT, int NT>
__global__ void edge_gemm3_kernel(const float* __restrict__ src, const int* __restrict__ idxp,
                                  const float* __restrict__ W, const float* __restrict__ sc,
                                  const float* __restrict__ bi, float* __restrict__ out,
                                  int M) {
    constexpr int CIN = 6;
    int wave = blockIdx.x * (blockDim.x >> 5) + (threadIdx.x >> 5);
    int lane = threadIdx.x & 31;
    constexpr int tilesN = COUT / (16 * NT);
    int tiles = (M >> 4) * tilesN;
    if (wave >= tiles) return;
    int tm = wave / tilesN, tg = wave % tilesN;
    int row0 = tm << 4, col0 = tg * 16 * NT;
    int half = lane >> 4, mr = lane & 15;

    int r = row0 + mr;
    int bn = r / KNN;
    int n = bn % NPTS, b = bn / NPTS;
    int nb = idxp[r];
    const float* ctr = src + ((size_t)b * NPTS + n) * 3;
    const float* nbr = src + ((size_t)b * NPTS + nb) * 3;

    float c0 = ctr[0], c1 = ctr[1], c2 = ctr[2];
    float d0 = nbr[0] - c0, d1 = nbr[1] - c1, d2 = nbr[2] - c2;
    bool h0 = (half == 0);   // only K=0..5 are non-zero; they live in half 0 elements 0..5
    v16bf af{};
    af[0] = (__bf16)(h0 ? d0 : 0.f);
    af[1] = (__bf16)(h0 ? d1 : 0.f);
    af[2] = (__bf16)(h0 ? d2 : 0.f);
    af[3] = (__bf16)(h0 ? c0 : 0.f);
    af[4] = (__bf16)(h0 ? c1 : 0.f);
    af[5] = (__bf16)(h0 ? c2 : 0.f);

    v8f acc[NT];
#pragma unroll
    for (int t = 0; t < NT; ++t) acc[t] = (v8f){};

#pragma unroll
    for (int t = 0; t < NT; ++t) {
        const float* wrow = W + (size_t)(col0 + t * 16 + mr) * CIN;
        float w0 = wrow[0], w1 = wrow[1], w2 = wrow[2];
        float w3 = wrow[3], w4 = wrow[4], w5 = wrow[5];
        v16bf bfv{};
        bfv[0] = (__bf16)(h0 ? w0 : 0.f);
        bfv[1] = (__bf16)(h0 ? w1 : 0.f);
        bfv[2] = (__bf16)(h0 ? w2 : 0.f);
        bfv[3] = (__bf16)(h0 ? w3 : 0.f);
        bfv[4] = (__bf16)(h0 ? w4 : 0.f);
        bfv[5] = (__bf16)(h0 ? w5 : 0.f);
        acc[t] = __builtin_amdgcn_wmma_f32_16x16x32_bf16(false, af, false, bfv,
                                                         (short)0, acc[t], false, false);
    }
    int mb = row0 + half * 8;
#pragma unroll
    for (int t = 0; t < NT; ++t) {
        int nc = col0 + t * 16 + (lane & 15);
        float s_ = sc[nc], b_ = bi[nc];
#pragma unroll
        for (int rr = 0; rr < 8; ++rr) {
            float z = acc[t][rr] * s_ + b_;
            z = z > 0.f ? z : NEGS * z;
            out[(size_t)(mb + rr) * COUT + nc] = z;
        }
    }
}

// =====================  max over k neighbors  =====================
__global__ void maxk_kernel(const float* __restrict__ h, float* __restrict__ out,
                            int rows, int C) {
    int i = blockIdx.x * blockDim.x + threadIdx.x;
    if (i >= rows * C) return;
    int c = i % C; int r = i / C;
    float m = -1e30f;
    for (int j = 0; j < KNN; ++j) m = fmaxf(m, h[((size_t)r * KNN + j) * C + c]);
    out[(size_t)r * C + c] = m;
}

// =====================  concat [x1|x2|x3] -> 192  =====================
__global__ void concat3_kernel(const float* __restrict__ x1, const float* __restrict__ x2,
                               const float* __restrict__ x3, float* __restrict__ out, int rows) {
    int i = blockIdx.x * blockDim.x + threadIdx.x;
    if (i >= rows * 192) return;
    int c = i % 192; int r = i / 192;
    float v = (c < 64) ? x1[(size_t)r * 64 + c]
            : (c < 128) ? x2[(size_t)r * 64 + (c - 64)]
                        : x3[(size_t)r * 64 + (c - 128)];
    out[i] = v;
}

// =====================  global max/mean over N  =====================
__global__ void gmaxmean_kernel(const float* __restrict__ x4, float* __restrict__ gmax,
                                float* __restrict__ gmean) {
    int i = blockIdx.x * blockDim.x + threadIdx.x;
    if (i >= NB * 1024) return;
    int c = i % 1024; int b = i / 1024;
    float m = -1e30f, s = 0.f;
    for (int n = 0; n < NPTS; ++n) {
        float v = x4[((size_t)b * NPTS + n) * 1024 + c];
        m = fmaxf(m, v); s += v;
    }
    gmax[i] = m; gmean[i] = s * (1.0f / NPTS);
}

// =====================  build h_in [B*N, 1216] = [g|x1|x2|x3]  =====================
__global__ void buildhin_kernel(const float* __restrict__ gmax, const float* __restrict__ x1,
                                const float* __restrict__ x2, const float* __restrict__ x3,
                                float* __restrict__ out) {
    size_t i = (size_t)blockIdx.x * blockDim.x + threadIdx.x;
    if (i >= (size_t)NB * NPTS * 1216) return;
    int c = (int)(i % 1216); size_t r = i / 1216; int b = (int)(r / NPTS);
    float v;
    if (c < 1024)      v = gmax[(size_t)b * 1024 + c];
    else if (c < 1088) v = x1[r * 64 + (c - 1024)];
    else if (c < 1152) v = x2[r * 64 + (c - 1088)];
    else               v = x3[r * 64 + (c - 1152)];
    out[i] = v;
}

// =====================  seg head: out[b,o,n] = h8[b,n,:] . w9[o,:]  =====================
__global__ void seg_kernel(const float* __restrict__ h8, const float* __restrict__ w9,
                           float* __restrict__ out) {
    int i = blockIdx.x * blockDim.x + threadIdx.x;
    if (i >= NB * 7 * NPTS) return;
    int n = i % NPTS; int rem = i / NPTS; int o = rem % 7; int b = rem / 7;
    const float* hr = h8 + ((size_t)b * NPTS + n) * 256;
    const float* wr = w9 + (size_t)o * 256;
    float z = 0.f;
    for (int c = 0; c < 256; ++c) z += hr[c] * wr[c];
    out[(size_t)b * 7 * NPTS + (size_t)o * NPTS + n] = z;
}

// =====================  y = [gmax | gmean]  =====================
__global__ void ycat_kernel(const float* __restrict__ gmax, const float* __restrict__ gmean,
                            float* __restrict__ out) {
    int i = blockIdx.x * blockDim.x + threadIdx.x;
    if (i >= NB * 2048) return;
    int c = i % 2048; int b = i / 2048;
    out[i] = (c < 1024) ? gmax[(size_t)b * 1024 + c] : gmean[(size_t)b * 1024 + (c - 1024)];
}

// =====================  small FC: z=(x.W^T(+wb))*s+b, leaky  =====================
__global__ void fc_kernel(const float* __restrict__ in, const float* __restrict__ W,
                          const float* __restrict__ wb, const float* __restrict__ sc,
                          const float* __restrict__ bb, float* __restrict__ out,
                          int CIN, int COUT, int act) {
    int i = blockIdx.x * blockDim.x + threadIdx.x;
    if (i >= NB * COUT) return;
    int o = i % COUT; int b = i / COUT;
    const float* row = in + (size_t)b * CIN;
    const float* wr = W + (size_t)o * CIN;
    float z = 0.f;
    for (int c = 0; c < CIN; ++c) z += row[c] * wr[c];
    if (wb) z += wb[o];
    if (sc) z = z * sc[o] + bb[o];
    if (act) z = z > 0.f ? z : NEGS * z;
    out[i] = z;
}

static inline int gemm_blocks(int M, int COUT, int NT) {
    int tiles = (M >> 4) * (COUT / (16 * NT));
    return (tiles * 32 + 255) / 256;
}

extern "C" void kernel_launch(void* const* d_in, const int* in_sizes, int n_in,
                              void* d_out, int out_size, void* d_ws, size_t ws_size,
                              hipStream_t stream) {
    const float* x   = (const float*)d_in[0];
    const float* w1  = (const float*)d_in[1];  const float* s1 = (const float*)d_in[2];  const float* b1 = (const float*)d_in[3];
    const float* w2  = (const float*)d_in[4];  const float* s2 = (const float*)d_in[5];  const float* b2 = (const float*)d_in[6];
    const float* w3  = (const float*)d_in[7];  const float* s3 = (const float*)d_in[8];  const float* b3 = (const float*)d_in[9];
    const float* w4  = (const float*)d_in[10]; const float* s4 = (const float*)d_in[11]; const float* b4 = (const float*)d_in[12];
    const float* w5  = (const float*)d_in[13]; const float* s5 = (const float*)d_in[14]; const float* b5 = (const float*)d_in[15];
    const float* w6  = (const float*)d_in[16]; const float* s6 = (const float*)d_in[17]; const float* b6 = (const float*)d_in[18];
    const float* w7  = (const float*)d_in[19]; const float* s7 = (const float*)d_in[20]; const float* b7 = (const float*)d_in[21];
    const float* w8  = (const float*)d_in[22]; const float* s8 = (const float*)d_in[23]; const float* b8 = (const float*)d_in[24];
    const float* w9  = (const float*)d_in[25];
    const float* l1w = (const float*)d_in[26]; const float* s9  = (const float*)d_in[27]; const float* b9  = (const float*)d_in[28];
    const float* l2w = (const float*)d_in[29]; const float* l2b = (const float*)d_in[30];
    const float* s10 = (const float*)d_in[31]; const float* b10 = (const float*)d_in[32];
    const float* l3w = (const float*)d_in[33]; const float* l3b = (const float*)d_in[34];

    const int M_edge = NB * NPTS * KNN;   // 327680
    const int M_pts  = NB * NPTS;         // 16384

    // -------- workspace layout (floats) --------
    float* ws = (float*)d_ws;
    size_t o = 0;
    float* xt    = ws + o; o += (size_t)M_pts * 3;
    int*   idx   = (int*)(ws + o); o += (size_t)M_edge;
    float* x1    = ws + o; o += (size_t)M_pts * 64;
    float* x2    = ws + o; o += (size_t)M_pts * 64;
    float* x3    = ws + o; o += (size_t)M_pts * 64;
    float* x4    = ws + o; o += (size_t)M_pts * 1024;
    float* gmx   = ws + o; o += (size_t)NB * 1024;
    float* gmn   = ws + o; o += (size_t)NB * 1024;
    float* bufA  = ws + o; o += (size_t)M_edge * 64;   // h1 / hin / ycat,y1,y2
    float* bufB  = ws + o; o += (size_t)M_edge * 64;   // h2 / cat123 / h7,h8
    float* h7    = bufB;
    float* h8    = bufB + (size_t)M_pts * 512;
    float* seg_out = (float*)d_out;
    float* y_out   = (float*)d_out + (size_t)NB * 7 * NPTS;

    dim3 knn_grid(NPTS / 256, NB);

    // x -> xt
    transpose_kernel<<<(NB * NPTS * 3 + 255) / 256, 256, 0, stream>>>(x, xt);

    // ---- EdgeConv block 1 ----
    knn_kernel<3><<<knn_grid, 256, 0, stream>>>(xt, idx);
    edge_gemm3_kernel<64, 4><<<gemm_blocks(M_edge, 64, 4), 256, 0, stream>>>(xt, idx, w1, s1, b1, bufA, M_edge);
    gemm_kernel<64, 64, 4><<<gemm_blocks(M_edge, 64, 4), 256, 0, stream>>>(bufA, w2, s2, b2, bufB, M_edge);
    maxk_kernel<<<(M_pts * 64 + 255) / 256, 256, 0, stream>>>(bufB, x1, M_pts, 64);

    // ---- EdgeConv block 2 ----
    knn_kernel<64><<<knn_grid, 256, 0, stream>>>(x1, idx);
    edge_gemm_kernel<64, 64, 4><<<gemm_blocks(M_edge, 64, 4), 256, 0, stream>>>(x1, idx, w3, s3, b3, bufA, M_edge);
    gemm_kernel<64, 64, 4><<<gemm_blocks(M_edge, 64, 4), 256, 0, stream>>>(bufA, w4, s4, b4, bufB, M_edge);
    maxk_kernel<<<(M_pts * 64 + 255) / 256, 256, 0, stream>>>(bufB, x2, M_pts, 64);

    // ---- EdgeConv block 3 ----
    knn_kernel<64><<<knn_grid, 256, 0, stream>>>(x2, idx);
    edge_gemm_kernel<64, 64, 4><<<gemm_blocks(M_edge, 64, 4), 256, 0, stream>>>(x2, idx, w5, s5, b5, bufA, M_edge);
    maxk_kernel<<<(M_pts * 64 + 255) / 256, 256, 0, stream>>>(bufA, x3, M_pts, 64);

    // ---- x4 = pw(cat123, w6) ----
    concat3_kernel<<<(M_pts * 192 + 255) / 256, 256, 0, stream>>>(x1, x2, x3, bufB, M_pts);
    gemm_kernel<192, 1024, 4><<<gemm_blocks(M_pts, 1024, 4), 256, 0, stream>>>(bufB, w6, s6, b6, x4, M_pts);

    // ---- global pooling ----
    gmaxmean_kernel<<<(NB * 1024 + 255) / 256, 256, 0, stream>>>(x4, gmx, gmn);

    // ---- segmentation branch ----
    {
        size_t tot = (size_t)M_pts * 1216;
        buildhin_kernel<<<(int)((tot + 255) / 256), 256, 0, stream>>>(gmx, x1, x2, x3, bufA);
    }
    gemm_kernel<1216, 512, 4><<<gemm_blocks(M_pts, 512, 4), 256, 0, stream>>>(bufA, w7, s7, b7, h7, M_pts);
    gemm_kernel<512, 256, 4><<<gemm_blocks(M_pts, 256, 4), 256, 0, stream>>>(h7, w8, s8, b8, h8, M_pts);
    seg_kernel<<<(NB * 7 * NPTS + 255) / 256, 256, 0, stream>>>(h8, w9, seg_out);

    // ---- classification branch ----
    float* ycat = bufA;                       // hin no longer needed
    float* y1   = bufA + (size_t)NB * 2048;
    float* y2   = y1   + (size_t)NB * 512;
    ycat_kernel<<<(NB * 2048 + 255) / 256, 256, 0, stream>>>(gmx, gmn, ycat);
    fc_kernel<<<(NB * 512 + 255) / 256, 256, 0, stream>>>(ycat, l1w, nullptr, s9,  b9,  y1, 2048, 512, 1);
    fc_kernel<<<(NB * 256 + 255) / 256, 256, 0, stream>>>(y1,   l2w, l2b,    s10, b10, y2, 512,  256, 1);
    fc_kernel<<<(NB * 5   + 255) / 256, 256, 0, stream>>>(y2,   l3w, l3b,    nullptr, nullptr, y_out, 256, 5, 0);
}
